// FBSNN_torch_1022202216498
// MI455X (gfx1250) — matrix operand order repaired and checked by reading, not compiled, not run on hardware
//
#include <hip/hip_runtime.h>

typedef _Float16 h16;
typedef __attribute__((ext_vector_type(16))) _Float16 v16h;
typedef __attribute__((ext_vector_type(8)))  float    v8f;

#define MM   4096
#define NNST 50
#define DD   100
#define HIDN 512
#define MT   32
#define R_RATE 0.05f
#define SIGMA  0.4f

// ---------------------------------------------------------------------------
// WMMA fragment index helpers (CDNA5 16x16x32 f16 layouts, CDNA5 ISA)
// A (16x32, MxK): lane = hsel*16 + (m&15); lanes<16 hold K={0..7,16..23},
// lanes>=16 hold K={8..15,24..31}; element e = (kk&7) + 8*(kk>>4), hsel=(kk>>3)&1.
// B (32x16, KxN): lane = hsel*16 + (n&15); lanes<16 hold K={0..7,16..23}, lanes>=16
// hold K={8..15,24..31}; element e = (kk&7) + 8*(kk>>4), as for A.
// Buffers are stored so each lane's 16 halves are contiguous (one 32B load).
// ---------------------------------------------------------------------------
__device__ __forceinline__ int packA_idx(int m, int k, int KT) {
  int rt = m >> 4;
  int kt = k >> 5;
  int kk = k & 31;
  int hs = (kk >> 3) & 1;
  int e  = (kk & 7) | ((kk >> 4) << 3);
  int ln = (hs << 4) | (m & 15);
  return ((rt * KT + kt) * 32 + ln) * 16 + e;
}

// ---------------------------------------------------------------------------
// Pack a weight matrix (optionally transposed, zero-padded) into the
// per-lane-contiguous B-fragment layout: dst[((kt*NT+nt)*32+lane)*16 + e].
// ---------------------------------------------------------------------------
__global__ void pack_weights(const float* __restrict__ src, h16* __restrict__ dst,
                             int Rs, int Cs, int Kp, int Np, int trans)
{
  const int NT = Np >> 4;
  const int total = (Kp >> 5) * NT * 32;
  const int i = blockIdx.x * blockDim.x + threadIdx.x;
  if (i >= total) return;
  const int lane = i & 31;
  const int tile = i >> 5;
  const int nt = tile % NT;
  const int kt = tile / NT;
  const int hsel = lane >> 4;
  const int n = nt * 16 + (lane & 15);
  h16* d = dst + (size_t)i * 16;
#pragma unroll
  for (int e = 0; e < 16; ++e) {
    const int k = kt * 32 + hsel * 8 + (e & 7) + ((e >> 3) << 4);
    float v = 0.f;
    if (trans) { if (n < Rs && k < Cs) v = src[(size_t)n * Cs + k]; }
    else       { if (k < Rs && n < Cs) v = src[(size_t)k * Cs + n]; }
    d[e] = (h16)v;
  }
}

__global__ void zero_out(float* out) {
  if (threadIdx.x == 0 && blockIdx.x == 0) out[0] = 0.f;
}

// ---------------------------------------------------------------------------
// One [MT x (16*NT)] = [MT x (32*KT)] @ B GEMM across the 8 waves of a block.
// Each wave owns one row-tile (rt = wave&1) and NT/4 column tiles, processed
// CW at a time: CW independent accumulator chains share one A fragment
// (no WMMA->WMMA hazard NOPs), and B fragments are double-buffered in
// registers so global loads for k-step kt+1 overlap the WMMAs of kt.
// The kt loop is kept rolled (unroll 1) so only 2*CW B fragments are live:
// acc(32) + 2xB(64) + A(8) ~ 120 VGPRs, no spilling.
// MODE 0: forward  -> acc init = bias; store z (f16), store sin(z) as packed A.
// MODE 1: backward -> g = acc * cos(z_stored); store g as packed A.
// MODE 2: input grad -> columns 1..100 are dL/dX -> Zout.
// ---------------------------------------------------------------------------
template<int CW>
__device__ __forceinline__ void loadB(v16h (&dst)[CW], const h16* __restrict__ Bp,
                                      int kt, int NT, int ct0, int lane)
{
#pragma unroll
  for (int c = 0; c < CW; ++c)
    dst[c] = *(const v16h*)(Bp + (((size_t)kt * NT + ct0 + c) * 32 + lane) * 16);
}

template<int MODE, int CW, int KT, int NT>
__device__ __forceinline__ void gemm_block(
    const h16* __restrict__ Ap,     // LDS, packed A fragments
    const h16* __restrict__ Bp,     // global, packed B fragments (L2 resident)
    const float* __restrict__ bias,
    h16* __restrict__ zst,
    h16* __restrict__ Aout,
    float* __restrict__ Zout)
{
  static_assert((KT & 1) == 0, "KT must be even");
  const int lane = threadIdx.x & 31;
  const int wave = threadIdx.x >> 5;
  const int hsel = lane >> 4;
  const int rt = wave & 1;
  constexpr int ctCount = NT >> 2;             // column tiles per wave
  const int ctBase = (wave >> 1) * ctCount;
  constexpr int ncols = NT * 16;
#pragma unroll 1
  for (int j = 0; j < ctCount; j += CW) {
    const int ct0 = ctBase + j;
    v8f acc[CW];
#pragma unroll
    for (int c = 0; c < CW; ++c) {
      float init = 0.f;
      if (MODE == 0) init = bias[(ct0 + c) * 16 + (lane & 15)];
#pragma unroll
      for (int r = 0; r < 8; ++r) acc[c][r] = init;
    }
    v16h bufA[CW], bufB[CW];
    loadB<CW>(bufA, Bp, 0, NT, ct0, lane);
#pragma unroll 1
    for (int kt = 0; kt < KT; kt += 2) {
      loadB<CW>(bufB, Bp, kt + 1, NT, ct0, lane);
      {
        v16h a = *(const v16h*)(Ap + ((rt * KT + kt) * 32 + lane) * 16);
#pragma unroll
        for (int c = 0; c < CW; ++c)
          acc[c] = __builtin_amdgcn_wmma_f32_16x16x32_f16(
              false, a, false, bufA[c], (short)0, acc[c], false, false);
      }
      loadB<CW>(bufA, Bp, (kt + 2 < KT) ? (kt + 2) : 0, NT, ct0, lane);
      {
        v16h a = *(const v16h*)(Ap + ((rt * KT + kt + 1) * 32 + lane) * 16);
#pragma unroll
        for (int c = 0; c < CW; ++c)
          acc[c] = __builtin_amdgcn_wmma_f32_16x16x32_f16(
              false, a, false, bufB[c], (short)0, acc[c], false, false);
      }
    }
#pragma unroll
    for (int c = 0; c < CW; ++c) {
      const int col = (ct0 + c) * 16 + (lane & 15);
#pragma unroll
      for (int r = 0; r < 8; ++r) {
        const int m = rt * 16 + r + 8 * hsel;
        const float v = acc[c][r];
        if (MODE == 0) {
          zst[m * ncols + col] = (h16)v;
          Aout[packA_idx(m, col, NT >> 1)] = (h16)__sinf(v);
        } else if (MODE == 1) {
          const float g = v * __cosf((float)zst[m * ncols + col]);
          Aout[packA_idx(m, col, NT >> 1)] = (h16)g;
        } else {
          if (col >= 1 && col <= DD) Zout[m * 104 + (col - 1)] = v;
        }
      }
    }
  }
}

// ---------------------------------------------------------------------------
// Persistent FBSNN kernel: 32 paths per workgroup, full 51-eval scan in LDS.
// ---------------------------------------------------------------------------
__global__ void __launch_bounds__(256)
fbsnn_kernel(const float* __restrict__ tg, const float* __restrict__ Wg,
             const float* __restrict__ Xi,
             const float* __restrict__ b0, const float* __restrict__ b1,
             const float* __restrict__ b2, const float* __restrict__ b3,
             const float* __restrict__ Wm4, const float* __restrict__ b4,
             const h16* __restrict__ W0p, const h16* __restrict__ W1p,
             const h16* __restrict__ W2p, const h16* __restrict__ W3p,
             const h16* __restrict__ W1T, const h16* __restrict__ W2T,
             const h16* __restrict__ W3T, const h16* __restrict__ W0T,
             float* __restrict__ out)
{
  extern __shared__ char smem[];
  h16* inbuf = (h16*)smem;              // 32*128  packed A, layer-0 input
  h16* abufA = inbuf + MT * 128;        // 32*512  activation ping (also grad A)
  h16* abufB = abufA + MT * HIDN;       // 32*512  activation pong
  h16* gB    = abufB + MT * HIDN;       // 32*512  gradient pong
  h16* zb0   = gB    + MT * HIDN;       // 4x 32*512 pre-activations
  h16* zb1   = zb0   + MT * HIDN;
  h16* zb2   = zb1   + MT * HIDN;
  h16* zb3   = zb2   + MT * HIDN;
  float* Xs  = (float*)(zb3 + MT * HIDN); // 32*104 path state X
  float* Zs  = Xs  + MT * 104;            // 32*104 gradient Z
  float* W4s = Zs  + MT * 104;            // 512
  float* bs0 = W4s + HIDN;                // 4x512 biases
  float* bs1 = bs0 + HIDN;
  float* bs2 = bs1 + HIDN;
  float* bs3 = bs2 + HIDN;
  float* ts1 = bs3 + HIDN;                // 32
  float* Yv  = ts1 + MT;                  // 32
  float* Ytl = Yv  + MT;                  // 32
  float* red = Ytl + MT;                  // 512
  float* red2 = red + 512;                // 256
  float* b4p  = red2 + 256;               // 1

  const int tid  = threadIdx.x;
  const int row  = tid >> 3;   // 0..31 : path within tile
  const int part = tid & 7;    // 8 threads per path for scalar work
  const int m0   = blockIdx.x * MT;
  const int gm   = m0 + row;

  for (int i = tid; i < HIDN; i += 256) {
    W4s[i] = Wm4[i];
    bs0[i] = b0[i]; bs1[i] = b1[i]; bs2[i] = b2[i]; bs3[i] = b3[i];
  }
  if (tid == 0) *b4p = b4[0];
  float accL = 0.f;
  __syncthreads();

  for (int n = 0; n <= NNST; ++n) {
    // ---- Euler step / state update (before overwriting Y,Z via net eval) ----
    const int dlo = part * 13;
    const int dhi = (dlo + 13 < DD) ? (dlo + 13) : DD;
    if (n == 0) {
      for (int d = dlo; d < dhi; ++d) Xs[row * 104 + d] = Xi[d];
      if (part == 0) ts1[row] = tg[(size_t)gm * (NNST + 1)];
    } else {
      float pXZ = 0.f, pZs = 0.f;
      const size_t bW1 = ((size_t)gm * (NNST + 1) + n) * DD;
      const size_t bW0 = bW1 - DD;
      for (int d = dlo; d < dhi; ++d) {
        const float dW = Wg[bW1 + d] - Wg[bW0 + d];
        const float x  = Xs[row * 104 + d];
        const float z  = Zs[row * 104 + d];
        const float sdw = SIGMA * x * dW;
        pXZ += x * z;
        pZs += z * sdw;
        Xs[row * 104 + d] = x + sdw;
      }
      red[(row * 8 + part) * 2 + 0] = pXZ;
      red[(row * 8 + part) * 2 + 1] = pZs;
      __syncthreads();
      if (part == 0) {
        float dXZ = 0.f, dZsum = 0.f;
        for (int q = 0; q < 8; ++q) {
          dXZ   += red[(row * 8 + q) * 2 + 0];
          dZsum += red[(row * 8 + q) * 2 + 1];
        }
        const float t1v = tg[(size_t)gm * (NNST + 1) + n];
        const float t0v = tg[(size_t)gm * (NNST + 1) + n - 1];
        const float y0  = Yv[row];
        const float phi = R_RATE * (y0 - dXZ);
        Ytl[row] = y0 + phi * (t1v - t0v) + dZsum;
        ts1[row] = t1v;
      }
    }
    __syncthreads();

    // ---- build layer-0 input [t, X, 0-pad] as packed A (K=128, KT=4) ----
    for (int i = tid; i < MT * 128; i += 256) {
      const int m = i >> 7, k = i & 127;
      const float v = (k == 0) ? ts1[m] : ((k <= DD) ? Xs[m * 104 + k - 1] : 0.f);
      inbuf[packA_idx(m, k, 4)] = (h16)v;
    }
    __syncthreads();

    // ---- forward MLP ----
    gemm_block<0,4, 4,32>(inbuf, W0p, bs0, zb0, abufA, nullptr); __syncthreads();
    gemm_block<0,4,16,32>(abufA, W1p, bs1, zb1, abufB, nullptr); __syncthreads();
    gemm_block<0,4,16,32>(abufB, W2p, bs2, zb2, abufA, nullptr); __syncthreads();
    gemm_block<0,4,16,32>(abufA, W3p, bs3, zb3, abufB, nullptr); __syncthreads();

    // ---- output head: u = s4 @ W4 + b4 (8 threads per row) ----
    {
      float p = 0.f;
      const int nlo = part * 64;
      for (int c = 0; c < 64; ++c) {
        const int col = nlo + c;
        p += (float)abufB[packA_idx(row, col, 16)] * W4s[col];
      }
      red[row * 8 + part] = p;
      __syncthreads();
      if (part == 0) {
        float s = 0.f;
        for (int q = 0; q < 8; ++q) s += red[row * 8 + q];
        Yv[row] = s + *b4p;
      }
    }
    __syncthreads();

    // ---- backward head: g_z4 = W4 * cos(z4) (grad of u wrt s4 is W4^T) ----
    for (int i = tid; i < MT * HIDN; i += 256) {
      const int m = i >> 9, c = i & 511;
      const float g = W4s[c] * __cosf((float)zb3[m * HIDN + c]);
      abufA[packA_idx(m, c, 16)] = (h16)g;
    }
    __syncthreads();

    // ---- backward GEMM chain ----
    gemm_block<1,4,16,32>(abufA, W3T, nullptr, zb2, gB,    nullptr); __syncthreads();
    gemm_block<1,4,16,32>(gB,    W2T, nullptr, zb1, abufA, nullptr); __syncthreads();
    gemm_block<1,4,16,32>(abufA, W1T, nullptr, zb0, gB,    nullptr); __syncthreads();
    gemm_block<2,2,16, 8>(gB,    W0T, nullptr, nullptr, nullptr, Zs); __syncthreads();

    // ---- step loss ----
    if (n >= 1 && part == 0) {
      const float d = Yv[row] - Ytl[row];
      accL += d * d;
    }
    __syncthreads();
  }

  // ---- terminal losses ----
  {
    float pG = 0.f, pZ2 = 0.f;
    const int dlo = part * 13;
    const int dhi = (dlo + 13 < DD) ? (dlo + 13) : DD;
    for (int d = dlo; d < dhi; ++d) {
      const float x = Xs[row * 104 + d];
      const float z = Zs[row * 104 + d];
      pG += x * x;
      const float e = z - 2.f * x;
      pZ2 += e * e;
    }
    red[(row * 8 + part) * 2 + 0] = pG;
    red[(row * 8 + part) * 2 + 1] = pZ2;
    __syncthreads();
    if (part == 0) {
      float gX = 0.f, sZ2 = 0.f;
      for (int q = 0; q < 8; ++q) {
        gX  += red[(row * 8 + q) * 2 + 0];
        sZ2 += red[(row * 8 + q) * 2 + 1];
      }
      const float d = Yv[row] - gX;
      accL += d * d + sZ2;
    }
  }
  red2[tid] = accL;
  __syncthreads();
  if (tid == 0) {
    float s = 0.f;
    for (int i = 0; i < 256; ++i) s += red2[i];
    atomicAdd(out, s * (1.0f / (float)MM));
  }
}

// ---------------------------------------------------------------------------
extern "C" void kernel_launch(void* const* d_in, const int* in_sizes, int n_in,
                              void* d_out, int out_size, void* d_ws, size_t ws_size,
                              hipStream_t stream)
{
  (void)in_sizes; (void)n_in; (void)out_size; (void)ws_size;
  const float* tg  = (const float*)d_in[0];
  const float* Wg  = (const float*)d_in[1];
  const float* Xi  = (const float*)d_in[2];
  const float* Wm0 = (const float*)d_in[3];
  const float* b0  = (const float*)d_in[4];
  const float* Wm1 = (const float*)d_in[5];
  const float* b1  = (const float*)d_in[6];
  const float* Wm2 = (const float*)d_in[7];
  const float* b2  = (const float*)d_in[8];
  const float* Wm3 = (const float*)d_in[9];
  const float* b3  = (const float*)d_in[10];
  const float* Wm4 = (const float*)d_in[11];
  const float* b4  = (const float*)d_in[12];

  h16* ws  = (h16*)d_ws;
  h16* W0p = ws;                      // 128*512
  h16* W1p = W0p + 128 * 512;         // 512*512 each below
  h16* W2p = W1p + 512 * 512;
  h16* W3p = W2p + 512 * 512;
  h16* W1T = W3p + 512 * 512;
  h16* W2T = W1T + 512 * 512;
  h16* W3T = W2T + 512 * 512;
  h16* W0T = W3T + 512 * 512;         // 512*128

  auto packLaunch = [&](const float* s, h16* dptr, int Rs, int Cs,
                        int Kp, int Np, int tr) {
    const int total = (Kp >> 5) * (Np >> 4) * 32;
    pack_weights<<<(total + 255) / 256, 256, 0, stream>>>(s, dptr, Rs, Cs, Kp, Np, tr);
  };
  packLaunch(Wm0, W0p, 101, 512, 128, 512, 0);
  packLaunch(Wm1, W1p, 512, 512, 512, 512, 0);
  packLaunch(Wm2, W2p, 512, 512, 512, 512, 0);
  packLaunch(Wm3, W3p, 512, 512, 512, 512, 0);
  packLaunch(Wm1, W1T, 512, 512, 512, 512, 1);
  packLaunch(Wm2, W2T, 512, 512, 512, 512, 1);
  packLaunch(Wm3, W3T, 512, 512, 512, 512, 1);
  packLaunch(Wm0, W0T, 101, 512, 512, 128, 1);

  zero_out<<<1, 32, 0, stream>>>((float*)d_out);

  // LDS: 118784 halves (237568 B) + 10081 floats (40324 B) = 277892 B (<320KB)
  const size_t ldsBytes =
      (size_t)(MT * 128 + 7 * MT * HIDN) * sizeof(h16) +
      (size_t)(2 * MT * 104 + 5 * HIDN + 3 * MT + 512 + 256 + 1) * sizeof(float);

  fbsnn_kernel<<<MM / MT, 256, ldsBytes, stream>>>(
      tg, Wg, Xi, b0, b1, b2, b3, Wm4, b4,
      W0p, W1p, W2p, W3p, W1T, W2T, W3T, W0T, (float*)d_out);
}
